// PAttention_77506979824242
// MI455X (gfx1250) — matrix operand order repaired and checked by reading, not compiled, not run on hardware
//
#include <hip/hip_runtime.h>
#include <hip/hip_bf16.h>

typedef __attribute__((ext_vector_type(16))) __bf16 v16bf;
typedef __attribute__((ext_vector_type(8)))  __bf16 v8bf;
typedef __attribute__((ext_vector_type(8)))  float  v8f;
typedef __attribute__((ext_vector_type(4)))  float  v4f;

#define D_DIM 1024
#define T_DIM 8192
#define L_DIM 8192
#define WIN   512
#define KC    1088   // padded per-tile key span: 16 + 2*511 + 32-align slop, mult of 32

__device__ __forceinline__ v16bf cat8(v8bf lo, v8bf hi) {
    v16bf r;
#pragma unroll
    for (int i = 0; i < 8; ++i) { r[i] = lo[i]; r[i + 8] = hi[i]; }
    return r;
}

// B fragment (32x16, KxN): lane n = lane&15; 16 contiguous bf16 along K at klbase
__device__ __forceinline__ v16bf load16(const __bf16* p) {
    return cat8(*(const v8bf*)p, *(const v8bf*)(p + 8));
}

// A fragment (16x32, MxK): lane m = lane&15; 16 contiguous bf16 along K at [2*akb..2*akb+15]
__device__ __forceinline__ v16bf loadA(const __bf16* rowk, int akb) {
    return cat8(*(const v8bf*)(rowk + 2 * akb), *(const v8bf*)(rowk + 2 * akb + 8));
}

__device__ __forceinline__ v8f wmma_bf16(v16bf a, v16bf b, v8f c) {
    return __builtin_amdgcn_wmma_f32_16x16x32_bf16(false, a, false, b, (short)0, c,
                                                   false, false);
}

// ---------------- Kernel 0: fp32 -> bf16 elementwise convert ----------------
__global__ __launch_bounds__(256) void cvt_bf16_kernel(const float* __restrict__ in,
                                                       __bf16* __restrict__ out) {
    int i = (blockIdx.x * 256 + threadIdx.x) * 4;
    v4f v = *(const v4f*)(in + i);
#pragma unroll
    for (int j = 0; j < 4; ++j) out[i + j] = (__bf16)v[j];
}

// ---------------- Kernel 1: transpose+convert V: VT[d][t] = (bf16)V[t][d] ----
__global__ __launch_bounds__(256) void transpose_v_kernel(const float* __restrict__ V,
                                                          __bf16* __restrict__ VT) {
    __shared__ __bf16 tile[32][33];
    int t0 = blockIdx.x * 32;
    int d0 = blockIdx.y * 32;
    int lx = threadIdx.x;        // 0..31
    int ly = threadIdx.y;        // 0..7
#pragma unroll
    for (int r = ly; r < 32; r += 8)
        tile[r][lx] = (__bf16)V[(t0 + r) * D_DIM + d0 + lx];
    __syncthreads();
#pragma unroll
    for (int r = ly; r < 32; r += 8)
        VT[(d0 + r) * T_DIM + t0 + lx] = tile[lx][r];
}

// ---------------- Kernel 2: Q = x @ Wq^T  (bf16, ping-pong pipelined) ----
// wave -> 16(M) x 64(N) tile; A fragment reused across 4 accumulators.
__global__ __launch_bounds__(256) void qgemm_kernel(const __bf16* __restrict__ xbf,
                                                    const __bf16* __restrict__ Wqbf,
                                                    __bf16* __restrict__ Qbf) {
    int gwave = (blockIdx.x * 256 + threadIdx.x) >> 5;
    int lane  = threadIdx.x & 31;
    int mtile = gwave >> 4;            // 0..511
    int nquad = gwave & 15;            // 0..15
    int m0 = mtile * 16, n0 = nquad * 64;
    int mn  = lane & 15;
    int akb = (lane < 16) ? 0 : 8;
    int bkb = (lane < 16) ? 0 : 16;

    const __bf16* arow = xbf + (size_t)(m0 + mn) * D_DIM;
    const __bf16* brow = Wqbf + (size_t)(n0 + mn) * D_DIM + bkb;   // +t*16 rows below

    v8f acc[4] = {v8f{}, v8f{}, v8f{}, v8f{}};
    // prologue: buffer A holds k=0
    v16bf aA = loadA(arow, akb);
    v16bf bA[4];
#pragma unroll
    for (int t = 0; t < 4; ++t) bA[t] = load16(brow + (size_t)t * 16 * D_DIM);

    // D/32 = 32 steps -> 16 even double-steps; both halves always valid.
    for (int k0 = 0; k0 < D_DIM; k0 += 64) {
        // load buffer B (k0+32)
        v16bf aB = loadA(arow + k0 + 32, akb);
        v16bf bB[4];
#pragma unroll
        for (int t = 0; t < 4; ++t)
            bB[t] = load16(brow + (size_t)t * 16 * D_DIM + k0 + 32);
        // compute on buffer A (k0)
#pragma unroll
        for (int t = 0; t < 4; ++t) acc[t] = wmma_bf16(aA, bA[t], acc[t]);
        // reload buffer A (k0+64)
        const bool more = (k0 + 64) < D_DIM;               // wave-uniform
        if (more) {
            aA = loadA(arow + k0 + 64, akb);
#pragma unroll
            for (int t = 0; t < 4; ++t)
                bA[t] = load16(brow + (size_t)t * 16 * D_DIM + k0 + 64);
        }
        // compute on buffer B (k0+32)
#pragma unroll
        for (int t = 0; t < 4; ++t) acc[t] = wmma_bf16(aB, bB[t], acc[t]);
    }
    int mrow = (lane < 16) ? 0 : 8;
#pragma unroll
    for (int t = 0; t < 4; ++t)
#pragma unroll
        for (int j = 0; j < 8; ++j)
            Qbf[(size_t)(m0 + mrow + j) * D_DIM + n0 + t * 16 + mn] = (__bf16)acc[t][j];
}

// ---------------- Kernel 3: fused banded attention ----------------
// One block (8 waves) per 16-query tile.
__global__ __launch_bounds__(256) void attn_kernel(const __bf16* __restrict__ Qbf,
                                                   const __bf16* __restrict__ Kbf,
                                                   const __bf16* __restrict__ VT,
                                                   float* __restrict__ out) {
    __shared__ __bf16 Pl[16][KC];   // scores, then probabilities (34 KB)
    __shared__ float  rsum[16];

    const int q0 = blockIdx.x * 16;
    int kb0 = q0 - (WIN - 1);
    if (kb0 < 0) kb0 = 0;
    kb0 &= ~31;                      // 32-aligned window start; kb0+KC covers band

    const int tid  = threadIdx.x;
    const int wave = tid >> 5;
    const int lane = tid & 31;
    const int nn   = lane & 15;
    const int akb  = (lane < 16) ? 0 : 8;
    const int bkb  = (lane < 16) ? 0 : 16;
    const int mrow = (lane < 16) ? 0 : 8;
    const float scale = 0.03125f;    // 1024^-0.5

    // ---- Phase 1: S = scale * Q K^T with band mask, stored bf16 in LDS ----
    // Key tiles paired (kt, kt+8): one A-fragment stream feeds two independent
    // WMMA accumulator chains; fragments ping-pong buffered across kd steps.
    const __bf16* qrow = Qbf + (size_t)(q0 + nn) * D_DIM;   // A row (lane&15)
    for (int kt = wave; kt < KC / 16; kt += 16) {
        const int  kt1  = kt + 8;
        const bool has1 = kt1 < KC / 16;                    // wave-uniform
        const int key0 = kb0 + kt * 16 + nn;
        const int key1 = kb0 + kt1 * 16 + nn;
        const int keyc0 = key0 < T_DIM ? key0 : T_DIM - 1;
        const int keyc1 = key1 < T_DIM ? key1 : T_DIM - 1;
        const __bf16* krow0 = Kbf + (size_t)keyc0 * D_DIM + bkb;
        const __bf16* krow1 = Kbf + (size_t)keyc1 * D_DIM + bkb;
        __builtin_prefetch(krow0, 0, 1);                    // global_prefetch_b8
        __builtin_prefetch(krow1, 0, 1);
        v8f c0 = {}, c1 = {};
        // prologue: buffer A holds kd=0
        v16bf aA  = loadA(qrow, akb);
        v16bf b0A = load16(krow0);
        v16bf b1A = load16(krow1);
        for (int kd = 0; kd < D_DIM; kd += 64) {
            // load buffer B (kd+32) -- always valid (32 steps, even count)
            v16bf aB  = loadA(qrow + kd + 32, akb);
            v16bf b0B = load16(krow0 + kd + 32);
            v16bf b1B = load16(krow1 + kd + 32);
            // compute on buffer A
            c0 = wmma_bf16(aA, b0A, c0);
            if (has1) c1 = wmma_bf16(aA, b1A, c1);
            // reload buffer A (kd+64)
            const bool more = (kd + 64) < D_DIM;            // wave-uniform
            if (more) {
                aA  = loadA(qrow + kd + 64, akb);
                b0A = load16(krow0 + kd + 64);
                b1A = load16(krow1 + kd + 64);
            }
            // compute on buffer B
            c0 = wmma_bf16(aB, b0B, c0);
            if (has1) c1 = wmma_bf16(aB, b1B, c1);
        }
#pragma unroll
        for (int j = 0; j < 8; ++j) {
            const int qq = q0 + mrow + j;
            {
                const int dqk = qq - key0;
                const bool valid = (dqk < WIN) && (dqk > -WIN) && (key0 < T_DIM);
                Pl[mrow + j][kt * 16 + nn] =
                    (__bf16)(valid ? c0[j] * scale : -__builtin_inff());
            }
            if (has1) {
                const int dqk = qq - key1;
                const bool valid = (dqk < WIN) && (dqk > -WIN) && (key1 < T_DIM);
                Pl[mrow + j][kt1 * 16 + nn] =
                    (__bf16)(valid ? c1[j] * scale : -__builtin_inff());
            }
        }
    }
    __syncthreads();

    // ---- Phase 2: row softmax (16 lanes per row, shuffle reductions) ----
    {
        const int row = tid >> 4, sub = tid & 15;
        float mx = -__builtin_inff();
        for (int c0 = sub; c0 < KC; c0 += 16) mx = fmaxf(mx, (float)Pl[row][c0]);
#pragma unroll
        for (int off = 8; off; off >>= 1) mx = fmaxf(mx, __shfl_xor(mx, off, 16));
        float sum = 0.f;
        for (int c0 = sub; c0 < KC; c0 += 16) {
            const float e = __expf((float)Pl[row][c0] - mx);
            sum += e;
            Pl[row][c0] = (__bf16)e;
        }
#pragma unroll
        for (int off = 8; off; off >>= 1) sum += __shfl_xor(sum, off, 16);
        if (sub == 0) rsum[row] = sum;
    }
    __syncthreads();

    // ---- Phase 3: O = P @ V ; wave owns 128-wide D slice; ping-pong ----
    {
        const int d0 = wave * 128;
        // chunk limit: cover keys [0, min(T-kb0, KC)), rounded up to 32
        int kclim = T_DIM - kb0;
        if (kclim > KC) kclim = KC;
        kclim = (kclim + 31) & ~31;

        const __bf16* vbase = VT + (size_t)(d0 + nn) * T_DIM;
        v8f acc[8] = {v8f{}, v8f{}, v8f{}, v8f{}, v8f{}, v8f{}, v8f{}, v8f{}};

        // prologue: buffer A holds kc=0
        v16bf aA = loadA(&Pl[nn][0], akb);
        v16bf bA[8];
        {
            int ks = kb0 + bkb;
            if (ks + 16 > T_DIM) ks = T_DIM - 16;
#pragma unroll
            for (int t = 0; t < 8; ++t)
                bA[t] = load16(vbase + (size_t)t * 16 * T_DIM + ks);
        }
        for (int kc = 0; kc < kclim; kc += 64) {
            const bool hasB = (kc + 32) < kclim;            // block-uniform
            v16bf aB, bB[8];
            if (hasB) {
                aB = loadA(&Pl[nn][kc + 32], akb);          // ds_load_b128 x2
                int ks = kb0 + kc + 32 + bkb;
                if (ks + 16 > T_DIM) ks = T_DIM - 16;       // clamp; masked P==0
#pragma unroll
                for (int t = 0; t < 8; ++t)
                    bB[t] = load16(vbase + (size_t)t * 16 * T_DIM + ks);
            }
            // compute on buffer A
#pragma unroll
            for (int t = 0; t < 8; ++t) acc[t] = wmma_bf16(aA, bA[t], acc[t]);
            // reload buffer A (kc+64)
            const bool moreA = (kc + 64) < kclim;           // block-uniform
            if (moreA) {
                aA = loadA(&Pl[nn][kc + 64], akb);
                int ks = kb0 + kc + 64 + bkb;
                if (ks + 16 > T_DIM) ks = T_DIM - 16;
#pragma unroll
                for (int t = 0; t < 8; ++t)
                    bA[t] = load16(vbase + (size_t)t * 16 * T_DIM + ks);
            }
            // compute on buffer B
            if (hasB) {
#pragma unroll
                for (int t = 0; t < 8; ++t) acc[t] = wmma_bf16(aB, bB[t], acc[t]);
            }
        }
#pragma unroll
        for (int j = 0; j < 8; ++j) {
            const float inv = 1.0f / rsum[mrow + j];
            float* orow = out + (size_t)(q0 + mrow + j) * D_DIM + d0;
#pragma unroll
            for (int t = 0; t < 8; ++t) orow[t * 16 + nn] = acc[t][j] * inv;
        }
    }
}

extern "C" void kernel_launch(void* const* d_in, const int* in_sizes, int n_in,
                              void* d_out, int out_size, void* d_ws, size_t ws_size,
                              hipStream_t stream) {
    (void)in_sizes; (void)n_in; (void)out_size; (void)ws_size;
    const float* x  = (const float*)d_in[0];
    const float* Wq = (const float*)d_in[1];
    const float* Pk = (const float*)d_in[2];
    const float* Pv = (const float*)d_in[3];
    float* out = (float*)d_out;

    __bf16* Qbf  = (__bf16*)d_ws;                        // 16 MB
    __bf16* Kbf  = Qbf  + (size_t)L_DIM * D_DIM;         // 16 MB
    __bf16* VT   = Kbf  + (size_t)T_DIM * D_DIM;         // 16 MB
    __bf16* xbf  = VT   + (size_t)T_DIM * D_DIM;         // 16 MB
    __bf16* Wqbf = xbf  + (size_t)L_DIM * D_DIM;         // 2 MB

    // fp32 -> bf16 conversions (K, x, Wq)
    cvt_bf16_kernel<<<(T_DIM * D_DIM) / (4 * 256), 256, 0, stream>>>(Pk, Kbf);
    cvt_bf16_kernel<<<(L_DIM * D_DIM) / (4 * 256), 256, 0, stream>>>(x, xbf);
    cvt_bf16_kernel<<<(D_DIM * D_DIM) / (4 * 256), 256, 0, stream>>>(Wq, Wqbf);
    // V -> bf16 transposed [D][T]
    transpose_v_kernel<<<dim3(T_DIM / 32, D_DIM / 32), dim3(32, 8), 0, stream>>>(Pv, VT);
    // Q = x @ Wq^T (bf16 in/out, f32 accumulate)
    qgemm_kernel<<<(L_DIM / 16) * (D_DIM / 64) / 8, 256, 0, stream>>>(xbf, Wqbf, Qbf);
    // fused banded attention
    attn_kernel<<<L_DIM / 16, 256, 0, stream>>>(Qbf, Kbf, VT, out);
}
